// SocialLSTM_66254165508743
// MI455X (gfx1250) — matrix operand order repaired and checked by reading, not compiled, not run on hardware
//
#include <hip/hip_runtime.h>
#include <hip/hip_bf16.h>
#include <stdint.h>

typedef __attribute__((ext_vector_type(16))) _Float16 v16h;
typedef __attribute__((ext_vector_type(8)))  _Float16 v8h;
typedef __attribute__((ext_vector_type(8)))  float    v8f;

#define H      128
#define EMB    32
#define G4H    512      // 4*H gate width
#define BT     32       // batch rows per block (2 WMMA M-tiles)
#define TOBS   8
#define TPRED  12
#define KROW   160      // xh row stride in halves: 32 pooled + 128 h
#define WMMA_F16(a, b, c) \
  __builtin_amdgcn_wmma_f32_16x16x32_f16(false, (a), false, (b), (short)0, (c), false, false)

// ---- LDS layout (bytes), dynamic shared memory ----
#define OFF_WHH   0         // f16 [512][128]  131072
#define OFF_A2    131072    // f16 [512][32]    32768
#define OFF_XH    163840    // f16 [32][160]    10240 (A operand: pooled|h)
#define OFF_GB    174080    // f32 [512]         2048 (obs-phase folded gate bias)
#define OFF_GBP   176128    // f32 [512]         2048 (pred-phase folded gate bias)
#define OFF_A1    178176    // f32 [512][2]      4096 (rank-2 obs/prev -> gates)
#define OFF_OBS   182272    // f32 [32][2]        256 (obs_t or prev disp)
#define OFF_IDX   182528    // u8  [32][64]      2048 (neighbor grid cells)
#define OFF_WOUT  184576    // f32 [2][128]      1024
#define OFF_BOUT  185600    // f32 [2]              8
#define LDS_BYTES 185664

// ---- offline weight folding: fold W_in / W_sp projections into gate weights ----
__global__ void social_lstm_pack(const float* __restrict__ W_in, const float* __restrict__ b_in,
                                 const float* __restrict__ W_sp, const float* __restrict__ b_sp,
                                 const float* __restrict__ W_ih, const float* __restrict__ W_hh,
                                 const float* __restrict__ b_ih, const float* __restrict__ b_hh,
                                 _Float16* __restrict__ A2, _Float16* __restrict__ Whh16,
                                 float* __restrict__ A1, float* __restrict__ gb,
                                 float* __restrict__ gbp) {
  int n = blockIdx.x * blockDim.x + threadIdx.x;
  if (n >= G4H) return;
  const float* wi = W_ih + (size_t)n * 256;   // [ :128]=agent cols, [128:256]=social cols
  float a0 = 0.f, a1 = 0.f, gin = 0.f, gsp = 0.f;
  for (int k = 0; k < H; ++k) {
    float wa = wi[k], ws = wi[128 + k];
    a0  += wa * W_in[k * 2 + 0];
    a1  += wa * W_in[k * 2 + 1];
    gin += wa * b_in[k];
    gsp += ws * b_sp[k];
  }
  A1[n * 2 + 0] = a0;
  A1[n * 2 + 1] = a1;
  for (int e = 0; e < EMB; ++e) {
    float s = 0.f;
    for (int k = 0; k < H; ++k) s += wi[128 + k] * W_sp[k * EMB + e];
    A2[n * EMB + e] = (_Float16)s;
  }
  for (int k = 0; k < H; ++k) Whh16[n * H + k] = (_Float16)W_hh[n * H + k];
  float bb = b_ih[n] + b_hh[n] + gin;
  gbp[n] = bb;        // pred phase: social input is zero
  gb[n]  = bb + gsp;  // obs phase: includes b_sp fold
}

// ---- persistent recurrent kernel: one block owns 32 batch rows for all 20 steps ----
__global__ void __launch_bounds__(256)
social_lstm_main(const float* __restrict__ obs, const float* __restrict__ nbr,
                 const float* __restrict__ emb, const float* __restrict__ Wout,
                 const float* __restrict__ bout, const _Float16* __restrict__ A2g,
                 const _Float16* __restrict__ Whhg, const float* __restrict__ A1g,
                 const float* __restrict__ gbg, const float* __restrict__ gbpg,
                 float* __restrict__ out) {
  extern __shared__ char smem[];
  _Float16* sWhh = (_Float16*)(smem + OFF_WHH);
  _Float16* sA2  = (_Float16*)(smem + OFF_A2);
  _Float16* sXH  = (_Float16*)(smem + OFF_XH);
  float*    sGB  = (float*)(smem + OFF_GB);
  float*    sGBP = (float*)(smem + OFF_GBP);
  float*    sA1  = (float*)(smem + OFF_A1);
  float*    sOBS = (float*)(smem + OFF_OBS);
  uint8_t*  sIDX = (uint8_t*)(smem + OFF_IDX);
  float*    sWOUT= (float*)(smem + OFF_WOUT);
  float*    sBOUT= (float*)(smem + OFF_BOUT);

  const int tid = threadIdx.x;
  const int b0  = blockIdx.x * BT;

  // Stage f16 weights + folded biases into LDS once (shared by all 20 steps).
  {
    const uint4* s0 = (const uint4*)Whhg; uint4* d0 = (uint4*)sWhh;
    for (int i = tid; i < (G4H * H * 2) / 16; i += 256) d0[i] = s0[i];
    const uint4* s1 = (const uint4*)A2g; uint4* d1 = (uint4*)sA2;
    for (int i = tid; i < (G4H * EMB * 2) / 16; i += 256) d1[i] = s1[i];
    for (int i = tid; i < G4H; i += 256) { sGB[i] = gbg[i]; sGBP[i] = gbpg[i]; }
    for (int i = tid; i < G4H * 2; i += 256) sA1[i] = A1g[i];
    if (tid < 256) sWOUT[tid] = Wout[tid];
    if (tid < 2)   sBOUT[tid] = bout[tid];
    for (int i = tid; i < BT * H; i += 256) {          // h0 = 0
      int r = i / H, k = i % H;
      sXH[r * KROW + 32 + k] = (_Float16)0.f;
    }
  }
  __syncthreads();

  const int w      = tid >> 5;     // wave32 id, owns hidden cols [16w,16w+16) of all 4 gates
  const int lane   = tid & 31;
  const int lane16 = lane & 15;
  const int laneHi = lane >> 4;

  v8f cst[2];                      // c-state lives in registers (C/D tile layout)
  #pragma unroll
  for (int m = 0; m < 2; ++m)
    #pragma unroll
    for (int v = 0; v < 8; ++v) cst[m][v] = 0.f;

  // ================= observation phase =================
  #pragma unroll 1
  for (int t = 0; t < TOBS; ++t) {
    // --- social pooling, phase A: grid-cell indices (only consumer of 67MB neighbors) ---
    {
      int row = tid >> 3, c8 = tid & 7;
      int b = b0 + row;
      float ox = obs[b * 16 + t * 2 + 0];
      float oy = obs[b * 16 + t * 2 + 1];
      if (c8 == 0) { sOBS[row * 2] = ox; sOBS[row * 2 + 1] = oy; }
      const float* nb = nbr + (size_t)b * 1024 + (size_t)(c8 * 8) * 16 + t * 2;
      #pragma unroll
      for (int j = 0; j < 8; ++j) {
        float rx = nb[j * 16 + 0] - ox;
        float ry = nb[j * 16 + 1] - oy;
        int ix = (int)floorf(rx * 2.0f);       // rel / CELL, CELL = 0.5
        int iy = (int)floorf(ry * 2.0f);
        ix = min(max(ix, -2), 1) + 2;
        iy = min(max(iy, -2), 1) + 2;
        sIDX[row * 64 + c8 * 8 + j] = (uint8_t)(ix * 4 + iy);
      }
    }
    __syncthreads();
    // --- phase B: gather-sum 2KB embed table (L1-resident), write f16 A-operand ---
    {
      int row = tid >> 3, e4 = (tid & 7) * 4;
      float s0 = 0.f, s1 = 0.f, s2 = 0.f, s3 = 0.f;
      #pragma unroll 4
      for (int n = 0; n < 64; ++n) {
        int f = sIDX[row * 64 + n];
        const float4 e = *(const float4*)(emb + f * 32 + e4);
        s0 += e.x; s1 += e.y; s2 += e.z; s3 += e.w;
      }
      _Float16* p = sXH + row * KROW + e4;
      p[0] = (_Float16)s0; p[1] = (_Float16)s1; p[2] = (_Float16)s2; p[3] = (_Float16)s3;
    }
    __syncthreads();

    // --- gate GEMM: acc = gb + obs@A1^T (rank-2 VALU) + [pooled|h]@[A2|Whh]^T (WMMA) ---
    float oxr[2][8], oyr[2][8];
    #pragma unroll
    for (int m = 0; m < 2; ++m)
      #pragma unroll
      for (int v = 0; v < 8; ++v) {
        int row = m * 16 + v + laneHi * 8;      // C/D layout row for (VGPR v, lane)
        oxr[m][v] = sOBS[row * 2 + 0];
        oyr[m][v] = sOBS[row * 2 + 1];
      }
    v8f acc[4][2];
    #pragma unroll
    for (int q = 0; q < 4; ++q) {               // i, f, g, o gate blocks
      int n = q * 128 + w * 16 + lane16;
      float a1x = sA1[n * 2], a1y = sA1[n * 2 + 1], bias = sGB[n];
      #pragma unroll
      for (int m = 0; m < 2; ++m)
        #pragma unroll
        for (int v = 0; v < 8; ++v)
          acc[q][m][v] = bias + oxr[m][v] * a1x + oyr[m][v] * a1y;
    }
    #pragma unroll
    for (int k = 0; k < 5; ++k) {               // k=0: pooled(A2), k=1..4: h(Whh)
      v16h a[2];
      #pragma unroll
      for (int m = 0; m < 2; ++m) {             // A as loaded here: 16 contiguous K halves at (L/16)*16
        const _Float16* p = sXH + (m * 16 + lane16) * KROW + k * 32 + laneHi * 16;
        union { v16h v; v8h h[2]; } u;
        u.h[0] = *(const v8h*)(p);
        u.h[1] = *(const v8h*)(p + 8);
        a[m] = u.v;
      }
      #pragma unroll
      for (int q = 0; q < 4; ++q) {             // B as loaded here: lane=col, 16 contiguous K halves (same k order as A)
        int n = q * 128 + w * 16 + lane16;
        const _Float16* bp = (k == 0) ? (sA2 + n * EMB + laneHi * 16)
                                      : (sWhh + n * H + (k - 1) * 32 + laneHi * 16);
        v16h bfr = *(const v16h*)bp;
        acc[q][0] = WMMA_F16(a[0], bfr, acc[q][0]);
        acc[q][1] = WMMA_F16(a[1], bfr, acc[q][1]);
      }
    }
    __syncthreads();                            // all A-operand reads done before h overwrite
    // --- LSTM nonlinearity (wave-local: i/f/g/o share hidden cols) + h writeback ---
    #pragma unroll
    for (int m = 0; m < 2; ++m)
      #pragma unroll
      for (int v = 0; v < 8; ++v) {
        float si = 1.f / (1.f + __expf(-acc[0][m][v]));
        float sf = 1.f / (1.f + __expf(-acc[1][m][v]));
        float so = 1.f / (1.f + __expf(-acc[3][m][v]));
        float c2 = sf * cst[m][v] + si * tanhf(acc[2][m][v]);
        float h2 = so * tanhf(c2);
        cst[m][v] = c2;
        int row = m * 16 + v + laneHi * 8;
        sXH[row * KROW + 32 + w * 16 + lane16] = (_Float16)h2;
      }
    __syncthreads();
  }

  // ================= prediction phase =================
  if (tid < 64) sOBS[tid] = 0.f;                // prev displacement = 0
  __syncthreads();
  #pragma unroll 1
  for (int t = 0; t < TPRED; ++t) {
    float oxr[2][8], oyr[2][8];
    #pragma unroll
    for (int m = 0; m < 2; ++m)
      #pragma unroll
      for (int v = 0; v < 8; ++v) {
        int row = m * 16 + v + laneHi * 8;
        oxr[m][v] = sOBS[row * 2 + 0];
        oyr[m][v] = sOBS[row * 2 + 1];
      }
    v8f acc[4][2];
    #pragma unroll
    for (int q = 0; q < 4; ++q) {
      int n = q * 128 + w * 16 + lane16;
      float a1x = sA1[n * 2], a1y = sA1[n * 2 + 1], bias = sGBP[n];
      #pragma unroll
      for (int m = 0; m < 2; ++m)
        #pragma unroll
        for (int v = 0; v < 8; ++v)
          acc[q][m][v] = bias + oxr[m][v] * a1x + oyr[m][v] * a1y;
    }
    #pragma unroll
    for (int k = 1; k < 5; ++k) {               // social input is zero in pred: skip k=0
      v16h a[2];
      #pragma unroll
      for (int m = 0; m < 2; ++m) {
        const _Float16* p = sXH + (m * 16 + lane16) * KROW + k * 32 + laneHi * 16;
        union { v16h v; v8h h[2]; } u;
        u.h[0] = *(const v8h*)(p);
        u.h[1] = *(const v8h*)(p + 8);
        a[m] = u.v;
      }
      #pragma unroll
      for (int q = 0; q < 4; ++q) {
        int n = q * 128 + w * 16 + lane16;
        const _Float16* bp = sWhh + n * H + (k - 1) * 32 + laneHi * 16;
        v16h bfr = *(const v16h*)bp;
        acc[q][0] = WMMA_F16(a[0], bfr, acc[q][0]);
        acc[q][1] = WMMA_F16(a[1], bfr, acc[q][1]);
      }
    }
    __syncthreads();
    #pragma unroll
    for (int m = 0; m < 2; ++m)
      #pragma unroll
      for (int v = 0; v < 8; ++v) {
        float si = 1.f / (1.f + __expf(-acc[0][m][v]));
        float sf = 1.f / (1.f + __expf(-acc[1][m][v]));
        float so = 1.f / (1.f + __expf(-acc[3][m][v]));
        float c2 = sf * cst[m][v] + si * tanhf(acc[2][m][v]);
        float h2 = so * tanhf(c2);
        cst[m][v] = c2;
        int row = m * 16 + v + laneHi * 8;
        sXH[row * KROW + 32 + w * 16 + lane16] = (_Float16)h2;
      }
    __syncthreads();
    // disp = h @ W_out^T + b_out  (tiny rank: 64 threads, 128-MAC dot each)
    if (tid < 64) {
      int row = tid >> 1, d = tid & 1;
      float s = sBOUT[d];
      const _Float16* hp = sXH + row * KROW + 32;
      const float* wp = sWOUT + d * H;
      #pragma unroll 8
      for (int k = 0; k < H; ++k) s += (float)hp[k] * wp[k];
      out[(size_t)(b0 + row) * (TPRED * 2) + t * 2 + d] = s;
      sOBS[row * 2 + d] = s;                    // becomes prev for next step
    }
    __syncthreads();
  }
}

extern "C" void kernel_launch(void* const* d_in, const int* in_sizes, int n_in,
                              void* d_out, int out_size, void* d_ws, size_t ws_size,
                              hipStream_t stream) {
  const float* obs   = (const float*)d_in[0];
  const float* nbr   = (const float*)d_in[1];
  const float* emb   = (const float*)d_in[2];
  const float* W_in  = (const float*)d_in[3];
  const float* b_in  = (const float*)d_in[4];
  const float* W_sp  = (const float*)d_in[5];
  const float* b_sp  = (const float*)d_in[6];
  const float* W_ih  = (const float*)d_in[7];
  const float* W_hh  = (const float*)d_in[8];
  const float* b_ih  = (const float*)d_in[9];
  const float* b_hh  = (const float*)d_in[10];
  const float* W_out = (const float*)d_in[11];
  const float* b_out = (const float*)d_in[12];
  float* out = (float*)d_out;

  char* ws = (char*)d_ws;
  _Float16* A2    = (_Float16*)(ws + 0);        //  32768 B
  _Float16* Whh16 = (_Float16*)(ws + 32768);    // 131072 B
  float*    A1    = (float*)(ws + 163840);      //   4096 B
  float*    gb    = (float*)(ws + 167936);      //   2048 B
  float*    gbp   = (float*)(ws + 169984);      //   2048 B

  const int B = in_sizes[0] / (TOBS * 2);       // 16384

  (void)hipFuncSetAttribute((const void*)social_lstm_main,
                            hipFuncAttributeMaxDynamicSharedMemorySize, LDS_BYTES);

  social_lstm_pack<<<2, 256, 0, stream>>>(W_in, b_in, W_sp, b_sp, W_ih, W_hh,
                                          b_ih, b_hh, A2, Whh16, A1, gb, gbp);
  social_lstm_main<<<B / BT, 256, LDS_BYTES, stream>>>(obs, nbr, emb, W_out, b_out,
                                                       A2, Whh16, A1, gb, gbp, out);
}
